// MLAAttention_89678917140584
// MI455X (gfx1250) — matrix operand order repaired and checked by reading, not compiled, not run on hardware
//
#include <hip/hip_runtime.h>
#include <hip/hip_bf16.h>

// Problem constants (match reference)
#define T_TOK   64
#define NH      128
#define DQK     192
#define DNOPE   128
#define DROPE   64
#define KVL     512
#define NTOPK   2048
#define SCACHE  65536
#define NHID    7168
#define SCALE_F (0.07216878364870323f) // 1/sqrt(192)

typedef __attribute__((ext_vector_type(16))) _Float16 v16h;
typedef __attribute__((ext_vector_type(8)))  float    v8f;
typedef unsigned int v4u __attribute__((ext_vector_type(4)));
typedef int          v8i __attribute__((ext_vector_type(8)));
typedef int          v4i __attribute__((ext_vector_type(4)));

// ---- Tensor Data Mover availability / arity detection -----------------------
#if defined(__has_builtin)
#  if __has_builtin(__builtin_amdgcn_tensor_load_to_lds)
#    define HAS_TDM 1
#  endif
#endif
#if __has_include(<hip/amd_detail/amd_gfx1250_TDM.h>)
#  define TDM_6ARG 1   // therock/clang-23 headers present -> 6-arg builtin
#endif

#ifdef HAS_TDM
__device__ __forceinline__ void tdm_load(v4u g0, v8i g1, v4i g2, v4i g3) {
#ifdef TDM_6ARG
  __builtin_amdgcn_tensor_load_to_lds(g0, g1, g2, g3, (v8i){0, 0, 0, 0, 0, 0, 0, 0}, 0);
#else
  __builtin_amdgcn_tensor_load_to_lds(g0, g1, g2, g3, 0);
#endif
}
#endif

// ---------------- WMMA fragment loaders (contiguous k = 16*(lane/16) + s order, same in A and B) -------
// A: 16x32 f16, element (m,k) at As[m*ld + k]; per-lane halves contiguous ->
// compiler merges to 2x ds_load_b128.
__device__ __forceinline__ v16h load_a_frag(const _Float16* As, int row0, int k0, int ld) {
  const int lane = threadIdx.x & 31;
  const int m = row0 + (lane & 15);
  const int kh = (lane >> 4) << 4; // 0 or 16
  v16h a;
#pragma unroll
  for (int v = 0; v < 8; ++v) {
    const int kb = (v << 1) + kh;
    a[2 * v]     = As[m * ld + k0 + kb];
    a[2 * v + 1] = As[m * ld + k0 + kb + 1];
  }
  return a;
}

// B (32x16): element (k,n) at Bs[k*ld + n]  (strided-column path; ds_load_u16)
__device__ __forceinline__ v16h load_b_frag(const _Float16* Bs, int k0, int n0, int ld) {
  const int lane = threadIdx.x & 31;
  const int n = n0 + (lane & 15);
  const int kh = (lane >> 4) << 4; // 0 or 16
  v16h b;
#pragma unroll
  for (int v = 0; v < 8; ++v) {
    const int k = k0 + kh + (v << 1);
    b[2 * v]     = Bs[k * ld + n];
    b[2 * v + 1] = Bs[(k + 1) * ld + n];
  }
  return b;
}

// B (32x16) from an [n][k] (transposed) LDS tile: contiguous -> ds_load_b128
__device__ __forceinline__ v16h load_bT_frag(const _Float16* Bs, int k0, int n0, int ldRow) {
  const int lane = threadIdx.x & 31;
  const int n = n0 + (lane & 15);
  const int kh = (lane >> 4) << 4;
  v16h b;
#pragma unroll
  for (int v = 0; v < 8; ++v) {
    const int k = k0 + kh + (v << 1);
    b[2 * v]     = Bs[n * ldRow + k];
    b[2 * v + 1] = Bs[n * ldRow + k + 1];
  }
  return b;
}

__device__ __forceinline__ v8f wmma_f16(v16h a, v16h b, v8f c) {
  return __builtin_amdgcn_wmma_f32_16x16x32_f16(false, a, false, b, (short)0, c, false, false);
}

// ---------------- f32 -> f16 bulk convert (once per launch, halves gather BW)
__global__ void cvt_f32_f16(const float* __restrict__ src, _Float16* __restrict__ dst) {
  const long i = ((long)blockIdx.x * blockDim.x + threadIdx.x) * 4;
  const float4 v = *(const float4*)(src + i);
  dst[i + 0] = (_Float16)v.x;
  dst[i + 1] = (_Float16)v.y;
  dst[i + 2] = (_Float16)v.z;
  dst[i + 3] = (_Float16)v.w;
}

// ---------------- Generic WMMA GEMM: C(MxN) = A(MxK) * B(KxN) ----------------
// B tile staged transposed in LDS so all frag reads are wide ds loads.
#define ALD 36
#define BLD 36
__global__ __launch_bounds__(128) void gemm_wmma(
    const float* __restrict__ A, long lda, long aZ,
    const float* __restrict__ B, long ldbk, long ldbn, long bZ,
    float* __restrict__ C, long ldc, long cZ,
    int M, int N, int K) {
  A += (long)blockIdx.z * aZ;
  B += (long)blockIdx.z * bZ;
  C += (long)blockIdx.z * cZ;
  __shared__ _Float16 As[32 * ALD];
  __shared__ _Float16 Bt[64 * BLD]; // [n][k]
  const int tid  = threadIdx.x;
  const int wave = tid >> 5;
  const int lane = tid & 31;
  const int mw = (wave >> 1) * 16;
  const int nw = (wave & 1) * 32;
  const int m0 = blockIdx.y * 32;
  const int n0 = blockIdx.x * 64;

  v8f acc0 = {}; v8f acc1 = {};
  for (int k0 = 0; k0 < K; k0 += 32) {
    __syncthreads();
#pragma unroll
    for (int i = 0; i < 8; ++i) {          // 32x32 A tile
      const int e = tid + i * 128;
      const int k = e & 31, m = e >> 5;
      As[m * ALD + k] = (_Float16)A[(long)(m0 + m) * lda + (k0 + k)];
    }
#pragma unroll
    for (int i = 0; i < 16; ++i) {         // 32x64 B tile, stored transposed
      const int e = tid + i * 128;
      const int n = e & 63, k = e >> 6;    // consecutive tid -> consecutive n
      Bt[n * BLD + k] = (_Float16)B[(long)(k0 + k) * ldbk + (long)(n0 + n) * ldbn];
    }
    __syncthreads();
    v16h a  = load_a_frag(As, mw, 0, ALD);
    v16h b0 = load_bT_frag(Bt, 0, nw, BLD);
    v16h b1 = load_bT_frag(Bt, 0, nw + 16, BLD);
    acc0 = wmma_f16(a, b0, acc0);
    acc1 = wmma_f16(a, b1, acc1);
  }
  const int nl = lane & 15;
  const int roff = (lane >> 4) * 8;
#pragma unroll
  for (int r = 0; r < 8; ++r) {
    const int m = m0 + mw + roff + r;
    C[(long)m * ldc + n0 + nw + nl]      = acc0[r];
    C[(long)m * ldc + n0 + nw + 16 + nl] = acc1[r];
  }
}

// ---------------- RoPE on q_pe (in place, q is (T,H,192), pe at +128) --------
__global__ void rope_kernel(float* __restrict__ q, const int* __restrict__ pos) {
  const int gid = blockIdx.x * blockDim.x + threadIdx.x; // T*H*32 threads
  const int i = gid & 31;
  const int h = (gid >> 5) & 127;
  const int t = gid >> 12;
  float* qp = q + ((long)(t * NH + h) * DQK + DNOPE);
  const float inv_freq = __powf(10000.0f, -((float)(2 * i)) / 64.0f);
  const float ang = (float)pos[t] * inv_freq;
  float s, c;
  __sincosf(ang, &s, &c);
  const float x1 = qp[i];
  const float x2 = qp[32 + i];
  qp[i]      = x1 * c - x2 * s;
  qp[32 + i] = x1 * s + x2 * c;
}

// ---------------- Flash attention over gathered top-k ------------------------
// WG = (token, 16-head group), 4 waves. KV chunk of 32 keys gathered by the
// Tensor Data Mover (gather-mode descriptors, 16 indices each) into LDS f16.
#define NC   32
#define KSLD 544   // 512 latent + 16-dword TDM pad -> 1088B row stride
__global__ __launch_bounds__(128) void attn_kernel(
    const float*    __restrict__ ql,    // (T,H,512)
    const float*    __restrict__ q,     // (T,H,192), rope at +128
    const _Float16* __restrict__ kvh,   // (S,512) f16
    const _Float16* __restrict__ kph,   // (S,64)  f16
    const int*      __restrict__ topk,  // (T,2048)
    float*          __restrict__ olat) {// (T,H,512)
  const int t  = blockIdx.x;
  const int hg = blockIdx.y;
  const int tid = threadIdx.x, wave = tid >> 5, lane = tid & 31;
  const long tb = (long)t * NTOPK;

  __shared__ _Float16 qs[16 * 576];        // q*SCALE, latent||rope packed
  __shared__ _Float16 ks[NC * KSLD];       // gathered kv latent rows
  __shared__ _Float16 ksp[NC * 64];        // gathered rope rows
  __shared__ float    spart[2][16 * 32];
  __shared__ _Float16 ps[16 * 32];
  __shared__ float    mrow[16], lrow[16], srow[16];

  for (int e = tid; e < 16 * 512; e += 128) {
    const int h = e >> 9, l = e & 511;
    qs[h * 576 + l] = (_Float16)(ql[((long)(t * NH + hg * 16 + h)) * KVL + l] * SCALE_F);
  }
  for (int e = tid; e < 16 * 64; e += 128) {
    const int h = e >> 6, d = e & 63;
    qs[h * 576 + 512 + d] =
        (_Float16)(q[((long)(t * NH + hg * 16 + h)) * DQK + DNOPE + d] * SCALE_F);
  }
  if (tid < 16) { mrow[tid] = -__builtin_inff(); lrow[tid] = 0.0f; }

  v8f o[8];
#pragma unroll
  for (int i = 0; i < 8; ++i) o[i] = (v8f){};

#ifdef HAS_TDM
  const unsigned long long kvA = (unsigned long long)kvh;
  const unsigned long long kpA = (unsigned long long)kph;
  const unsigned ksb  = (unsigned)(unsigned long long)(void*)&ks[0];
  const unsigned kspb = (unsigned)(unsigned long long)(void*)&ksp[0];
#endif

  for (int c0 = 0; c0 < NTOPK; c0 += NC) {
    __syncthreads();
#ifdef HAS_TDM
    if (wave == 0) {
      // pack 32 clipped 16-bit gather indices (uniform scalar loads)
      unsigned ipk[16];
#pragma unroll
      for (int j = 0; j < 16; ++j) {
        const int i0 = topk[tb + c0 + 2 * j];
        const int i1 = topk[tb + c0 + 2 * j + 1];
        const unsigned u0 = (i0 < 0) ? 0u : ((unsigned)i0 & 0xFFFFu);
        const unsigned u1 = (i1 < 0) ? 0u : ((unsigned)i1 & 0xFFFFu);
        ipk[j] = u0 | (u1 << 16);
      }
#pragma unroll
      for (int d = 0; d < 2; ++d) {
        const v4i g2 = (v4i){(int)ipk[d * 8 + 0], (int)ipk[d * 8 + 1],
                             (int)ipk[d * 8 + 2], (int)ipk[d * 8 + 3]};
        const v4i g3 = (v4i){(int)ipk[d * 8 + 4], (int)ipk[d * 8 + 5],
                             (int)ipk[d * 8 + 6], (int)ipk[d * 8 + 7]};
        // --- kv latent rows: 512 x f16, padded to 544/row in LDS ---
        {
          const v4u g0 = (v4u){0x80000001u,                       // gather_mode|count=1
                               ksb + (unsigned)(d * 16 * KSLD * 2),
                               (unsigned)(kvA & 0xFFFFFFFFu),
                               (unsigned)((kvA >> 32) & 0x1FFFFFFu) | (2u << 30)};
          const v8i g1 = (v8i){
              (int)((1u << 16) | (1u << 20) | (7u << 22) | (15u << 25)), // 2B, pad 16dw/256dw
              (int)(512u << 16),          // tensor_dim0 = 512
              0,                          // dim0 hi, dim1 lo16 (65536 -> 0)
              (int)(1u | (512u << 16)),   // dim1 hi16 = 1, tile_dim0 = 512
              16,                         // tile_dim1 = 16 gather indices
              512, 0, 0};                 // dim0_stride = 512
          tdm_load(g0, g1, g2, g3);
        }
        // --- rope rows: 64 x f16, no pad ---
        {
          const v4u g0 = (v4u){0x80000001u,
                               kspb + (unsigned)(d * 16 * 64 * 2),
                               (unsigned)(kpA & 0xFFFFFFFFu),
                               (unsigned)((kpA >> 32) & 0x1FFFFFFu) | (2u << 30)};
          const v8i g1 = (v8i){(int)(1u << 16),
                               (int)(64u << 16),
                               0,
                               (int)(1u | (64u << 16)),
                               16,
                               64, 0, 0};
          tdm_load(g0, g1, g2, g3);
        }
      }
      __builtin_amdgcn_s_wait_tensorcnt(0);
    }
#else
    { // fallback: plain vector gather of f16 rows
      const int key = tid >> 2, q4 = tid & 3;
      const int j = topk[tb + c0 + key];
      const int idx = (j < 0) ? 0 : j;
      const uint4* s4 = (const uint4*)(kvh + (long)idx * KVL) + q4 * 16;
      uint4* dv = (uint4*)(&ks[key * KSLD + q4 * 128]);
#pragma unroll
      for (int i = 0; i < 16; ++i) dv[i] = s4[i];
      const uint4* p4 = (const uint4*)(kph + (long)idx * DROPE) + q4 * 2;
      uint4* dp = (uint4*)(&ksp[key * 64 + q4 * 16]);
      dp[0] = p4[0]; dp[1] = p4[1];
    }
#endif
    __syncthreads();
    // ---- S = q.k over 576 packed features; 2 key-tiles x 2 K-halves --------
    {
      v8f sc = {};
      const int nt = wave >> 1;
      const int kh = (wave & 1) * 9;
#pragma unroll
      for (int s = 0; s < 9; ++s) {
        const int step = kh + s;
        const int kk = step * 32;
        v16h a = load_a_frag(qs, 0, kk, 576);
        v16h b = (step < 16) ? load_bT_frag(ks, kk, nt * 16, KSLD)
                             : load_bT_frag(ksp, kk - 512, nt * 16, 64);
        sc = wmma_f16(a, b, sc);
      }
      const int nl = lane & 15, roff = (lane >> 4) * 8;
#pragma unroll
      for (int r = 0; r < 8; ++r)
        spart[wave & 1][(roff + r) * 32 + nt * 16 + nl] = sc[r];
    }
    __syncthreads();
    // ---- online softmax (one thread per head row) --------------------------
    if (tid < 16) {
      const int h = tid;
      const float mold = mrow[h];
      float mx = mold;
#pragma unroll 4
      for (int j = 0; j < NC; ++j) {
        const int jj = topk[tb + c0 + j];
        float s = spart[0][h * 32 + j] + spart[1][h * 32 + j];
        if (jj < 0) s = -__builtin_inff();
        spart[0][h * 32 + j] = s;
        if (s > mx) mx = s;
      }
      const float sf = (mold == mx) ? 1.0f : __expf(mold - mx);
      float l = lrow[h] * sf;
#pragma unroll 4
      for (int j = 0; j < NC; ++j) {
        const float s = spart[0][h * 32 + j];
        const float p = (s == -__builtin_inff()) ? 0.0f : __expf(s - mx);
        ps[h * 32 + j] = (_Float16)p;
        l += p;
      }
      mrow[h] = mx; lrow[h] = l; srow[h] = sf;
    }
    __syncthreads();
    // ---- O = O*sf + P @ V (each wave owns 128 latent cols) -----------------
    {
      v16h a = load_a_frag(ps, 0, 0, 32);
      const int roff = (lane >> 4) * 8;
#pragma unroll
      for (int nt = 0; nt < 8; ++nt) {
#pragma unroll
        for (int r = 0; r < 8; ++r) o[nt][r] *= srow[roff + r];
        v16h b = load_b_frag(ks, 0, wave * 128 + nt * 16, KSLD);
        o[nt] = wmma_f16(a, b, o[nt]);
      }
    }
  }
  __syncthreads();
  // ---- normalize and write o_latent ----
  {
    const int nl = lane & 15, roff = (lane >> 4) * 8;
#pragma unroll
    for (int nt = 0; nt < 8; ++nt) {
#pragma unroll
      for (int r = 0; r < 8; ++r) {
        const int h = hg * 16 + roff + r;
        const float inv = 1.0f / lrow[roff + r];
        olat[((long)(t * NH + h)) * KVL + wave * 128 + nt * 16 + nl] = o[nt][r] * inv;
      }
    }
  }
}

// ---------------- launch ------------------------------------------------------
extern "C" void kernel_launch(void* const* d_in, const int* in_sizes, int n_in,
                              void* d_out, int out_size, void* d_ws, size_t ws_size,
                              hipStream_t stream) {
  const float* x    = (const float*)d_in[0];
  const float* wq   = (const float*)d_in[1];
  const float* wkvb = (const float*)d_in[2];
  const float* wo   = (const float*)d_in[3];
  const float* kvc  = (const float*)d_in[4];
  const float* kpe  = (const float*)d_in[5];
  const int*   pos  = (const int*)d_in[6];
  const int*   topk = (const int*)d_in[7];
  float* out = (float*)d_out;

  float* ws   = (float*)d_ws;
  float* q    = ws;                                 // (T,H,192)
  float* qln  = q    + (size_t)T_TOK * NH * DQK;    // (T,H,512)
  float* olat = qln  + (size_t)T_TOK * NH * KVL;    // (T,H,512)
  float* o    = olat + (size_t)T_TOK * NH * KVL;    // (T,H*128)
  _Float16* kvh = (_Float16*)(o + (size_t)T_TOK * NH * DNOPE); // (S,512) f16
  _Float16* kph = kvh + (size_t)SCACHE * KVL;                  // (S,64)  f16

  const dim3 blk(128);

  // 0) one-time f16 conversion of the caches (halves gather HBM traffic)
  cvt_f32_f16<<<dim3((SCACHE * (size_t)KVL) / (256 * 4)), dim3(256), 0, stream>>>(kvc, kvh);
  cvt_f32_f16<<<dim3((SCACHE * (size_t)DROPE) / (256 * 4)), dim3(256), 0, stream>>>(kpe, kph);

  // 1) q = x @ wq : (64 x 1536) x (1536 x 24576)
  gemm_wmma<<<dim3((NH * DQK) / 64, T_TOK / 32, 1), blk, 0, stream>>>(
      x, 1536, 0, wq, (long)NH * DQK, 1, 0, q, (long)NH * DQK, 0,
      T_TOK, NH * DQK, 1536);

  // 2) RoPE on q_pe
  rope_kernel<<<dim3((T_TOK * NH * 32) / 256), dim3(256), 0, stream>>>(q, pos);

  // 3) ql_nope[t,h,l] = sum_n q_nope[t,h,n] * w_kv_b[l,h,n]  (per-head GEMM)
  gemm_wmma<<<dim3(KVL / 64, T_TOK / 32, NH), blk, 0, stream>>>(
      q, (long)NH * DQK, DQK,
      wkvb, 1, (long)NH * 256, 256,
      qln, (long)NH * KVL, KVL,
      T_TOK, KVL, DNOPE);

  // 4) flash attention over top-k TDM gathers
  attn_kernel<<<dim3(T_TOK, NH / 16), blk, 0, stream>>>(qln, q, kvh, kph, topk, olat);

  // 5) o[t, h*128+v] = sum_l olat[t,h,l] * w_kv_b[l,h,128+v]  (per-head GEMM)
  gemm_wmma<<<dim3(DNOPE / 64, T_TOK / 32, NH), blk, 0, stream>>>(
      olat, (long)NH * KVL, KVL,
      wkvb + DNOPE, (long)NH * 256, 1, 256,
      o, (long)NH * DNOPE, DNOPE,
      T_TOK, DNOPE, KVL);

  // 6) out = o @ wo : (64 x 16384) x (16384 x 7168)
  gemm_wmma<<<dim3(NHID / 64, T_TOK / 32, 1), blk, 0, stream>>>(
      o, (long)NH * DNOPE, 0, wo, NHID, 1, 0, out, NHID, 0,
      T_TOK, NHID, (long)NH * DNOPE);
}
